// ASSA_38482906972648
// MI455X (gfx1250) — matrix operand order repaired and checked by reading, not compiled, not run on hardware
//
#include <hip/hip_runtime.h>
#include <math.h>

typedef _Float16 h16;
typedef __attribute__((ext_vector_type(16))) _Float16 v16h;
typedef __attribute__((ext_vector_type(8)))  _Float16 v8h;
typedef __attribute__((ext_vector_type(8)))  float    v8f;
typedef int v4i __attribute__((vector_size(16)));

#define EPSF 1e-8f

// -------- async global->LDS copy path (gfx1250), with safe fallback --------
#if defined(__AMDGCN__) && __has_builtin(__builtin_amdgcn_global_load_async_to_lds_b128)
#define USE_ASYNC 1
#else
#define USE_ASYNC 0
#endif

#if USE_ASYNC
#define ASYNC_CP16(gsrc, ldst)                                          \
  __builtin_amdgcn_global_load_async_to_lds_b128(                       \
      (v4i*)(void*)(gsrc), (v4i*)(void*)(ldst), 0, 0)
#if __has_builtin(__builtin_amdgcn_s_wait_asynccnt)
#define WAIT_ASYNC(n) __builtin_amdgcn_s_wait_asynccnt(n)
#else
#define WAIT_ASYNC(n) asm volatile("s_wait_asynccnt %0" ::"i"(n))
#endif
#else
#define ASYNC_CP16(gsrc, ldst) (*(v8h*)(ldst) = *(const v8h*)(gsrc))
#define WAIT_ASYNC(n)
#endif

// ---------- WMMA helpers (CDNA5 wave32, 16x16x32 f16 -> f32) ----------

__device__ __forceinline__ v8f wmma_f16(v16h a, v16h b, v8f c) {
  // (neg_a, A, neg_b, B, c_mod, C, reuse_a, reuse_b)
  return __builtin_amdgcn_wmma_f32_16x16x32_f16(false, a, false, b, (short)0, c,
                                                false, false);
}

// A-fragment: 16x32 f16 tile from a row-major matrix (ld in elements).
// Lane L (L<16): row L, K = 0..7 (v0..3) and 16..23 (v4..7).
// Lane L+16:     row L, K = 8..15 and 24..31.
__device__ __forceinline__ v16h load_a_frag(const h16* base, int ld) {
  const int lane = threadIdx.x & 31;
  const h16* p = base + (lane & 15) * ld + ((lane >> 4) << 3);
  v8h lo = *(const v8h*)p;
  v8h hi = *(const v8h*)(p + 16);
  return __builtin_shufflevector(lo, hi, 0, 1, 2, 3, 4, 5, 6, 7,
                                 8, 9, 10, 11, 12, 13, 14, 15);
}

// B-fragment: 32x16 f16 operand stored column-major, i.e. from Bt (N rows x K
// cols) row-major with leading dim ld. Lane n (n<16): col n, K=0..7 and
// 16..23; lane n+16: col n, K=8..15 and 24..31.
__device__ __forceinline__ v16h load_b_frag(const h16* base, int ld) {
  const int lane = threadIdx.x & 31;
  const h16* p = base + (lane & 15) * ld + ((lane >> 4) << 3);
  v8h lo = *(const v8h*)p;
  v8h hi = *(const v8h*)(p + 16);
  return __builtin_shufflevector(lo, hi, 0, 1, 2, 3, 4, 5, 6, 7,
                                 8, 9, 10, 11, 12, 13, 14, 15);
}

// 16-lane (half-wave) reductions matching C/D layout rows.
__device__ __forceinline__ float row_sum16(float v) {
  v += __shfl_xor(v, 1, 32);
  v += __shfl_xor(v, 2, 32);
  v += __shfl_xor(v, 4, 32);
  v += __shfl_xor(v, 8, 32);
  return v;
}
__device__ __forceinline__ float row_max16(float v) {
  v = fmaxf(v, __shfl_xor(v, 1, 32));
  v = fmaxf(v, __shfl_xor(v, 2, 32));
  v = fmaxf(v, __shfl_xor(v, 4, 32));
  v = fmaxf(v, __shfl_xor(v, 8, 32));
  return v;
}

// ---------- prep kernels ----------

__global__ void cvt_f32_to_f16(const float* __restrict__ src,
                               h16* __restrict__ dst, int n) {
  int i = (blockIdx.x * blockDim.x + threadIdx.x) * 4;
  if (i + 3 < n) {
    float4 v = *(const float4*)(src + i);
    dst[i + 0] = (h16)v.x; dst[i + 1] = (h16)v.y;
    dst[i + 2] = (h16)v.z; dst[i + 3] = (h16)v.w;
  }
}

// src: K x N row-major f32 ; dst: N x K row-major f16 (transposed)
__global__ void transpose_f32_to_f16(const float* __restrict__ src,
                                     h16* __restrict__ dst, int K, int N) {
  int i = blockIdx.x * blockDim.x + threadIdx.x;
  if (i < K * N) {
    int n = i / K, k = i % K;
    dst[n * K + k] = (h16)src[k * N + n];
  }
}

// ---------- QKV GEMM: (8192x512) @ (512x1536), scatter into Q/K/Vt ----------
// Q,K layout: (B*H, N, D) row-major f16.  Vt layout: (B*H, D, N) row-major f16.

__global__ __launch_bounds__(256) void qkv_gemm_kernel(
    const h16* __restrict__ xh, const h16* __restrict__ wt,
    h16* __restrict__ Q, h16* __restrict__ Kb, h16* __restrict__ Vt) {
  const int wid = threadIdx.x >> 5;
  const int w = blockIdx.x * 8 + wid;
  const int NT = 1536 / 64;              // 24 column tiles
  const int mt = w / NT, nt = w % NT;
  const int m0 = mt * 16, n0 = nt * 64;
  const int lane = threadIdx.x & 31;
  const int hf = lane >> 4, nl = lane & 15;

  v8f acc[4];
#pragma unroll
  for (int j = 0; j < 4; ++j) acc[j] = (v8f){};

  for (int kk = 0; kk < 512; kk += 32) {
    v16h a = load_a_frag(xh + m0 * 512 + kk, 512);
#pragma unroll
    for (int j = 0; j < 4; ++j) {
      v16h b = load_b_frag(wt + (n0 + 16 * j) * 512 + kk, 512);
      acc[j] = wmma_f16(a, b, acc[j]);
    }
  }

  const int which = n0 / 512;            // 0=q 1=k 2=v (tile never straddles)
  const int c0 = n0 % 512;
  const int h = c0 / 64;                 // 64-wide tile == one full head
  const int bidx = m0 / 2048;
  const int nn0 = m0 % 2048;
  const int bh = bidx * 8 + h;

  if (which < 2) {
    h16* dst = (which == 0 ? Q : Kb) + (size_t)bh * 2048 * 64;
#pragma unroll
    for (int j = 0; j < 4; ++j)
#pragma unroll
      for (int g = 0; g < 8; ++g)
        dst[(nn0 + g + 8 * hf) * 64 + 16 * j + nl] = (h16)acc[j][g];
  } else {
    h16* dst = Vt + (size_t)bh * 64 * 2048;
#pragma unroll
    for (int j = 0; j < 4; ++j)
#pragma unroll
      for (int g = 0; g < 8; ++g)
        dst[(16 * j + nl) * 2048 + nn0 + g + 8 * hf] = (h16)acc[j][g];
  }
}

// ---------- fused two-branch flash attention ----------
// grid (N/128, B*H), 8 waves/block, each wave owns a 16-query tile.
// K/V tiles for the whole block are double-buffered in LDS via async copies.

__global__ __launch_bounds__(256) void attn_kernel(
    const h16* __restrict__ Q, const h16* __restrict__ Kb,
    const h16* __restrict__ Vt, const float* __restrict__ alpha,
    const float* __restrict__ beta, h16* __restrict__ out) {
  __shared__ alignas(64) h16 kbuf[2][32 * 64];  // 32 keys x 64 dims
  __shared__ alignas(64) h16 vbuf[2][64 * 32];  // 64 dims x 32 keys (V^T)
  __shared__ alignas(64) h16 psp[8][16][32];    // relu^2 probs per wave
  __shared__ alignas(64) h16 pdn[8][16][32];    // exp probs per wave

  const int tid = threadIdx.x;
  const int wid = tid >> 5;
  const int lane = tid & 31;
  const int hf = lane >> 4, nl = lane & 15;
  const int bh = blockIdx.y;
  const int b = bh >> 3, h = bh & 7;
  const int q0 = blockIdx.x * 128 + wid * 16;

  const h16* Qb = Q + (size_t)bh * 2048 * 64;
  const h16* Kh = Kb + (size_t)bh * 2048 * 64;
  const h16* Vh = Vt + (size_t)bh * 64 * 2048;

  // Q fragments with softmax scale folded in (1/8 is exact in f16).
  v16h qa0 = load_a_frag(Qb + q0 * 64 + 0, 64);
  v16h qa1 = load_a_frag(Qb + q0 * 64 + 32, 64);
  const h16 sc = (h16)0.125f;
#pragma unroll
  for (int i = 0; i < 16; ++i) { qa0[i] *= sc; qa1[i] *= sc; }

  // Cooperative tile staging: K tile = 32 rows x 128B, V tile = 64 rows x 64B.
  const int kr = tid >> 3, kc = (tid & 7) * 8;   // K: 8 chunks/row
  const int vr = tid >> 2, vc = (tid & 3) * 8;   // V: 4 chunks/row

  float m_run[8], l_run[8], ssum[8];
  v8f acc_d[4], acc_s[4];
#pragma unroll
  for (int g = 0; g < 8; ++g) {
    m_run[g] = -__builtin_inff(); l_run[g] = 0.f; ssum[g] = 0.f;
  }
#pragma unroll
  for (int j = 0; j < 4; ++j) { acc_d[j] = (v8f){}; acc_s[j] = (v8f){}; }

  // Prologue: stage tile 0 into buffer 0 (2 async ops per wave).
  ASYNC_CP16(Kh + kr * 64 + kc, &kbuf[0][kr * 64 + kc]);
  ASYNC_CP16(Vh + vr * 2048 + vc, &vbuf[0][vr * 32 + vc]);

  int cb = 0;
  for (int kt = 0; kt < 2048; kt += 32) {
    // Stage next tile into the other buffer, then wait for current tile.
    if (kt + 32 < 2048) {
      const int nb = cb ^ 1;
      ASYNC_CP16(Kh + (kt + 32 + kr) * 64 + kc, &kbuf[nb][kr * 64 + kc]);
      ASYNC_CP16(Vh + vr * 2048 + (kt + 32) + vc, &vbuf[nb][vr * 32 + vc]);
      WAIT_ASYNC(2);   // in-order: the 2 older (current-tile) ops are done
    } else {
      WAIT_ASYNC(0);
    }
    __syncthreads();

    // ---- scores: 16 queries x 32 keys, B-fragments from LDS ----
    v8f zc = (v8f){};
    v16h kb00 = load_b_frag(&kbuf[cb][0 * 64 + 0], 64);
    v16h kb01 = load_b_frag(&kbuf[cb][0 * 64 + 32], 64);
    v8f s0 = wmma_f16(qa0, kb00, zc);
    s0 = wmma_f16(qa1, kb01, s0);
    v16h kb10 = load_b_frag(&kbuf[cb][16 * 64 + 0], 64);
    v16h kb11 = load_b_frag(&kbuf[cb][16 * 64 + 32], 64);
    v8f s1 = wmma_f16(qa0, kb10, zc);
    s1 = wmma_f16(qa1, kb11, s1);

    // ---- online stats + probability tiles ----
    float fs[8];
#pragma unroll
    for (int g = 0; g < 8; ++g) {
      float mx = row_max16(fmaxf(s0[g], s1[g]));
      float tm = fmaxf(m_run[g], mx);
      fs[g] = __expf(m_run[g] - tm);
      float e0 = __expf(s0[g] - tm);
      float e1 = __expf(s1[g] - tm);
      float r0 = s0[g] > 0.f ? s0[g] * s0[g] : 0.f;
      float r1 = s1[g] > 0.f ? s1[g] * s1[g] : 0.f;
      int m = g + 8 * hf;
      pdn[wid][m][nl]      = (h16)e0;
      pdn[wid][m][16 + nl] = (h16)e1;
      psp[wid][m][nl]      = (h16)r0;
      psp[wid][m][16 + nl] = (h16)r1;
      l_run[g] = l_run[g] * fs[g] + row_sum16(e0 + e1);
      ssum[g] += row_sum16(r0 + r1);
      m_run[g] = tm;
    }
#pragma unroll
    for (int j = 0; j < 4; ++j)
#pragma unroll
      for (int g = 0; g < 8; ++g) acc_d[j][g] *= fs[g];

    __syncthreads();   // order prob-tile stores before cross-lane frag loads

    v16h pa_s = load_a_frag(&psp[wid][0][0], 32);
    v16h pa_d = load_a_frag(&pdn[wid][0][0], 32);
#pragma unroll
    for (int j = 0; j < 4; ++j) {
      v16h vb = load_b_frag(&vbuf[cb][(16 * j) * 32], 32);
      acc_s[j] = wmma_f16(pa_s, vb, acc_s[j]);
      acc_d[j] = wmma_f16(pa_d, vb, acc_d[j]);
    }
    __syncthreads();   // all reads done before next stage overwrites buffers
    cb ^= 1;
  }

  const float aw = 1.f / (1.f + __expf(-alpha[0]));
  const float bw = 1.f / (1.f + __expf(-beta[0]));
  const float wsum = aw + bw + EPSF;
  const float ca = aw / wsum, cb2 = bw / wsum;

  // out layout: (B, N, H*D) f16 — matches the reference transpose+reshape.
  h16* orow = out + ((size_t)(b * 2048 + q0)) * 512 + h * 64;
#pragma unroll
  for (int j = 0; j < 4; ++j)
#pragma unroll
    for (int g = 0; g < 8; ++g) {
      float o = ca * acc_s[j][g] / (ssum[g] + EPSF) +
                cb2 * acc_d[j][g] / l_run[g];
      orow[(g + 8 * hf) * 512 + 16 * j + nl] = (h16)o;
    }
}

// ---------- output projection: (8192x512) @ (512x512) + bias -> f32 ----------

__global__ __launch_bounds__(256) void proj_kernel(
    const h16* __restrict__ A, const h16* __restrict__ Wt,
    const float* __restrict__ bias, float* __restrict__ out) {
  const int wid = threadIdx.x >> 5;
  const int w = blockIdx.x * 8 + wid;
  const int NT = 512 / 64;               // 8 column tiles
  const int mt = w / NT, nt = w % NT;
  const int m0 = mt * 16, n0 = nt * 64;
  const int lane = threadIdx.x & 31;
  const int hf = lane >> 4, nl = lane & 15;

  v8f acc[4];
#pragma unroll
  for (int j = 0; j < 4; ++j) acc[j] = (v8f){};

  for (int kk = 0; kk < 512; kk += 32) {
    v16h a = load_a_frag(A + m0 * 512 + kk, 512);
#pragma unroll
    for (int j = 0; j < 4; ++j) {
      v16h b = load_b_frag(Wt + (n0 + 16 * j) * 512 + kk, 512);
      acc[j] = wmma_f16(a, b, acc[j]);
    }
  }
#pragma unroll
  for (int j = 0; j < 4; ++j) {
    float bj = bias[n0 + 16 * j + nl];
#pragma unroll
    for (int g = 0; g < 8; ++g)
      out[(size_t)(m0 + g + 8 * hf) * 512 + n0 + 16 * j + nl] = acc[j][g] + bj;
  }
}

// ---------- launch ----------

extern "C" void kernel_launch(void* const* d_in, const int* in_sizes, int n_in,
                              void* d_out, int out_size, void* d_ws,
                              size_t ws_size, hipStream_t stream) {
  (void)in_sizes; (void)n_in; (void)out_size; (void)ws_size;
  const float* x      = (const float*)d_in[0];
  const float* w_qkv  = (const float*)d_in[1];
  const float* alpha  = (const float*)d_in[2];
  const float* beta   = (const float*)d_in[3];
  const float* w_proj = (const float*)d_in[4];
  const float* b_proj = (const float*)d_in[5];
  float* out = (float*)d_out;

  char* ws = (char*)d_ws;
  size_t off = 0;
  auto alloc = [&](size_t bytes) -> char* {
    char* p = ws + off;
    off += (bytes + 255) & ~(size_t)255;
    return p;
  };
  h16* xh  = (h16*)alloc(4ull * 2048 * 512 * 2);    // x in f16
  h16* wqt = (h16*)alloc(1536ull * 512 * 2);        // w_qkv^T f16
  h16* wpt = (h16*)alloc(512ull * 512 * 2);         // w_proj^T f16
  h16* Q   = (h16*)alloc(32ull * 2048 * 64 * 2);    // (BH,N,D)
  h16* Kb  = (h16*)alloc(32ull * 2048 * 64 * 2);    // (BH,N,D)
  h16* Vt  = (h16*)alloc(32ull * 64 * 2048 * 2);    // (BH,D,N)
  h16* ao  = (h16*)alloc(4ull * 2048 * 512 * 2);    // attention output (B,N,C)

  cvt_f32_to_f16<<<4096, 256, 0, stream>>>(x, xh, 4 * 2048 * 512);
  transpose_f32_to_f16<<<(512 * 1536 + 255) / 256, 256, 0, stream>>>(
      w_qkv, wqt, 512, 1536);
  transpose_f32_to_f16<<<(512 * 512 + 255) / 256, 256, 0, stream>>>(
      w_proj, wpt, 512, 512);
  qkv_gemm_kernel<<<1536, 256, 0, stream>>>(xh, wqt, Q, Kb, Vt);
  attn_kernel<<<dim3(16, 32), 256, 0, stream>>>(Q, Kb, Vt, alpha, beta, ao);
  proj_kernel<<<512, 256, 0, stream>>>(ao, wpt, b_proj, out);
}
